// _GraphConv_65137474011776
// MI455X (gfx1250) — matrix operand order repaired and checked by reading, not compiled, not run on hardware
//
#include <hip/hip_runtime.h>
#include <hip/hip_bf16.h>

// ---------------------------------------------------------------------------
// GNN-LF propagation on MI455X (gfx1250), wave32 + V_WMMA_F32_16X16X32_F16.
//
// Roofline: adj (8192^2 f32 = 268MB) dominates. Batches packed into the
// feature dim (width 64) -> adj read once per iteration; adj converted to f16
// once -> 268+134MB convert + 4x134MB stream ~= 0.95GB ~= 41us @ 23.3TB/s.
// hkT (1MB, f16, transposed) stays L2-resident as the B operand.
//
// prop kernel: one wave owns a 32-row x 64-feature block (2 A tiles, 8 f32
// accumulators), K-split 8 ways -> 2048 waves. Inner loop is double-buffered:
// step s+1's A/B tiles load into fresh registers while step s's 8 WMMAs issue,
// so s_wait_loadcnt is nonzero and HBM latency is hidden inside the wave.
// ---------------------------------------------------------------------------

#define NN   8192
#define DIN  64
#define DOUT 32
#define CF   64            // packed feature width = 2 batches * 32
#define KP   8             // K-split factor for the big GEMM
#define KLEN (NN / KP)     // 1024
#define MB   32            // rows per wave (two 16-row A tiles)

typedef __attribute__((ext_vector_type(16))) _Float16 v16h;
typedef __attribute__((ext_vector_type(8)))  _Float16 v8h;
typedef __attribute__((ext_vector_type(4)))  _Float16 v4h;
typedef __attribute__((ext_vector_type(8)))  float    v8f;

// A tile (16x32 f16): lane(r,hi): row = r, elems0-15 = K[hi*16..+16]
// -> two contiguous b128 loads.
static __device__ __forceinline__ v16h loadA(const _Float16* p) {
  const v8h* q = (const v8h*)p;
  v8h a0 = q[0], a1 = q[1];
  return __builtin_shufflevector(a0, a1, 0,1,2,3,4,5,6,7,8,9,10,11,12,13,14,15);
}
// B tile (32x16 f16): lane(r,hi): col = r, elems0-15 = K[hi*16..+16]
// -> two contiguous b128 loads.
static __device__ __forceinline__ v16h loadB(const _Float16* p) {
  const v8h* q = (const v8h*)p;
  v8h b0 = q[0], b1 = q[1];
  return __builtin_shufflevector(b0, b1, 0,1,2,3,4,5,6,7,8,9,10,11,12,13,14,15);
}

// ---------------------------------------------------------------- convert ---
__global__ __launch_bounds__(256) void k_convert(const float* __restrict__ adj,
                                                 _Float16* __restrict__ adjh) {
  long t = (long)blockIdx.x * 256 + threadIdx.x;   // 16M threads, 4 elems each
  float4 v = ((const float4*)adj)[t];
  v4h o = { (_Float16)v.x, (_Float16)v.y, (_Float16)v.z, (_Float16)v.w };
  ((v4h*)adjh)[t] = o;
}

// ------------------------------------------------------------------- proj ---
// thread t -> (n, c): c = b*32 + f.  h -> hkT (f16, transposed), h0 -> h0f (f32)
__global__ __launch_bounds__(256) void k_proj(const float* __restrict__ x,
                                              const float* __restrict__ x0,
                                              const float* __restrict__ W,
                                              const float* __restrict__ W0,
                                              _Float16* __restrict__ hkT,
                                              float* __restrict__ h0f) {
  __shared__ float sW[DIN * DOUT];
  __shared__ float sW0[DIN * DOUT];
  for (int i = threadIdx.x; i < DIN * DOUT; i += 256) { sW[i] = W[i]; sW0[i] = W0[i]; }
  __syncthreads();

  long t = (long)blockIdx.x * 256 + threadIdx.x;   // 8192*64 threads
  long n = t >> 6;
  int  c = (int)(t & 63);
  int  b = c >> 5, f = c & 31;

  const float* xr  = x  + ((long)b * NN + n) * DIN;
  const float* x0r = x0 + ((long)b * NN + n) * DIN;
  float h = 0.f, h0 = 0.f;
#pragma unroll
  for (int d = 0; d < DIN; ++d) {
    h  = fmaf(xr[d],  sW[d * DOUT + f],  h);
    h0 = fmaf(x0r[d], sW0[d * DOUT + f], h0);
  }
  h0f[n * CF + c]       = h0;
  hkT[(long)c * NN + n] = (_Float16)h;   // hk(0) = h
}

// ------------------------------------------------------------------- prop ---
__global__ __launch_bounds__(256) void k_prop(const _Float16* __restrict__ adjh,
                                              const _Float16* __restrict__ hkT,
                                              float* __restrict__ part) {
  const int wave = blockIdx.x * 8 + (threadIdx.x >> 5);  // 2048 waves total
  const int lane = threadIdx.x & 31;
  const int rb   = wave & 255;       // 32-row block index (256 blocks)
  const int kp   = wave >> 8;        // K partition (8)
  const int r    = lane & 15;
  const int hi   = lane >> 4;

  const long rowBase = (long)rb * MB;
  const long kBase   = (long)kp * KLEN;

  const _Float16* a0p = adjh + (rowBase + r) * (long)NN + kBase + (long)hi * 16;
  const _Float16* a1p = a0p + 16L * NN;                       // rows 16..31
  const _Float16* bp  = hkT + (long)r * NN + kBase + (long)hi * 16;
  const long BT = 16L * NN;                                   // feature-tile stride

  v8f acc00 = {}, acc01 = {}, acc02 = {}, acc03 = {};
  v8f acc10 = {}, acc11 = {}, acc12 = {}, acc13 = {};

  // prologue: load step 0
  v16h A0 = loadA(a0p), A1 = loadA(a1p);
  v16h B0 = loadB(bp), B1 = loadB(bp + BT), B2 = loadB(bp + 2 * BT), B3 = loadB(bp + 3 * BT);

  const int STEPS = KLEN / 32;       // 32
#pragma unroll 2
  for (int s = 0; s < STEPS - 1; ++s) {
    const int kn = (s + 1) * 32;
    __builtin_prefetch(a0p + kn + 256, 0, 3);   // stream adj ~8 steps ahead
    __builtin_prefetch(a1p + kn + 256, 0, 3);
    // load step s+1 into fresh registers (overlaps with this step's WMMAs)
    v16h A0n = loadA(a0p + kn), A1n = loadA(a1p + kn);
    v16h B0n = loadB(bp + kn),          B1n = loadB(bp + kn + BT);
    v16h B2n = loadB(bp + kn + 2 * BT), B3n = loadB(bp + kn + 3 * BT);

    acc00 = __builtin_amdgcn_wmma_f32_16x16x32_f16(false, A0, false, B0, (short)0, acc00, false, false);
    acc01 = __builtin_amdgcn_wmma_f32_16x16x32_f16(false, A0, false, B1, (short)0, acc01, false, false);
    acc02 = __builtin_amdgcn_wmma_f32_16x16x32_f16(false, A0, false, B2, (short)0, acc02, false, false);
    acc03 = __builtin_amdgcn_wmma_f32_16x16x32_f16(false, A0, false, B3, (short)0, acc03, false, false);
    acc10 = __builtin_amdgcn_wmma_f32_16x16x32_f16(false, A1, false, B0, (short)0, acc10, false, false);
    acc11 = __builtin_amdgcn_wmma_f32_16x16x32_f16(false, A1, false, B1, (short)0, acc11, false, false);
    acc12 = __builtin_amdgcn_wmma_f32_16x16x32_f16(false, A1, false, B2, (short)0, acc12, false, false);
    acc13 = __builtin_amdgcn_wmma_f32_16x16x32_f16(false, A1, false, B3, (short)0, acc13, false, false);

    A0 = A0n; A1 = A1n; B0 = B0n; B1 = B1n; B2 = B2n; B3 = B3n;
  }
  // epilogue: final step's WMMAs
  acc00 = __builtin_amdgcn_wmma_f32_16x16x32_f16(false, A0, false, B0, (short)0, acc00, false, false);
  acc01 = __builtin_amdgcn_wmma_f32_16x16x32_f16(false, A0, false, B1, (short)0, acc01, false, false);
  acc02 = __builtin_amdgcn_wmma_f32_16x16x32_f16(false, A0, false, B2, (short)0, acc02, false, false);
  acc03 = __builtin_amdgcn_wmma_f32_16x16x32_f16(false, A0, false, B3, (short)0, acc03, false, false);
  acc10 = __builtin_amdgcn_wmma_f32_16x16x32_f16(false, A1, false, B0, (short)0, acc10, false, false);
  acc11 = __builtin_amdgcn_wmma_f32_16x16x32_f16(false, A1, false, B1, (short)0, acc11, false, false);
  acc12 = __builtin_amdgcn_wmma_f32_16x16x32_f16(false, A1, false, B2, (short)0, acc12, false, false);
  acc13 = __builtin_amdgcn_wmma_f32_16x16x32_f16(false, A1, false, B3, (short)0, acc13, false, false);

  // D layout: vgpr v, lane(r,hi) -> row m = tile*16 + v + 8*hi, col = ft*16 + r
  float* pr = part + ((long)kp * NN + rowBase) * CF;
#pragma unroll
  for (int v = 0; v < 8; ++v) {
    long ro = (long)(v + 8 * hi) * CF + r;
    pr[ro + 0 * 16] = acc00[v];
    pr[ro + 1 * 16] = acc01[v];
    pr[ro + 2 * 16] = acc02[v];
    pr[ro + 3 * 16] = acc03[v];
    long r1 = ro + 16L * CF;
    pr[r1 + 0 * 16] = acc10[v];
    pr[r1 + 1 * 16] = acc11[v];
    pr[r1 + 2 * 16] = acc12[v];
    pr[r1 + 3 * 16] = acc13[v];
  }
}

// ---------------------------------------------------------- reduce_update ---
__global__ __launch_bounds__(256) void k_reduce_update(const float* __restrict__ part,
                                                       const float* __restrict__ h0f,
                                                       float* __restrict__ outIter,
                                                       _Float16* __restrict__ hkT) {
  long t = (long)blockIdx.x * 256 + threadIdx.x;   // 8192*64
  long n = t >> 6;
  int  c = (int)(t & 63);
  const long S = (long)NN * CF;
  float y = 0.f;
#pragma unroll
  for (int kp = 0; kp < KP; ++kp) y += part[t + kp * S];
  float hn = 0.9f * y + 0.1f * h0f[t];
  outIter[t]            = hn;            // f32 for BN
  hkT[(long)c * NN + n] = (_Float16)hn;  // f16 transposed for next iteration
}

// ---------------------------------------------------------------- BN stats --
__global__ __launch_bounds__(128) void k_bn_partial(const float* __restrict__ outs,
                                                    float* __restrict__ psum,
                                                    float* __restrict__ psq) {
  int cc = blockIdx.x;            // channel 0..127 (iter*32 + f)
  int chunk = blockIdx.y;         // 0..63, 128 nodes each
  int iter = cc >> 5, f = cc & 31;
  long n = (long)chunk * 128 + threadIdx.x;
  const float* base = outs + (long)iter * NN * CF + n * CF;
  float v0 = base[f], v1 = base[32 + f];          // both batches
  __shared__ float ss[128], sq[128];
  ss[threadIdx.x] = v0 + v1;
  sq[threadIdx.x] = v0 * v0 + v1 * v1;
  __syncthreads();
  for (int st = 64; st > 0; st >>= 1) {
    if ((int)threadIdx.x < st) { ss[threadIdx.x] += ss[threadIdx.x + st]; sq[threadIdx.x] += sq[threadIdx.x + st]; }
    __syncthreads();
  }
  if (threadIdx.x == 0) { psum[cc * 64 + chunk] = ss[0]; psq[cc * 64 + chunk] = sq[0]; }
}

__global__ __launch_bounds__(128) void k_bn_final(const float* __restrict__ psum,
                                                  const float* __restrict__ psq,
                                                  const float* __restrict__ gamma,
                                                  const float* __restrict__ beta,
                                                  float* __restrict__ scale,
                                                  float* __restrict__ bias) {
  int cc = threadIdx.x;           // 128 channels
  float s = 0.f, q = 0.f;
  for (int i = 0; i < 64; ++i) { s += psum[cc * 64 + i]; q += psq[cc * 64 + i]; }
  const float M = 16384.0f;       // B*N samples
  float mean = s / M;
  float var  = q / M - mean * mean;          // biased, as torch BN training
  float sc   = gamma[cc] * rsqrtf(var + 1e-5f);
  scale[cc] = sc;
  bias[cc]  = beta[cc] - mean * sc;
}

// --------------------------------------------------------------- finalize ---
__global__ __launch_bounds__(256) void k_finalize(const float* __restrict__ outs,
                                                  const float* __restrict__ scale,
                                                  const float* __restrict__ bias,
                                                  float* __restrict__ out) {
  long t = (long)blockIdx.x * 256 + threadIdx.x;  // 2*8192*128
  int  cc = (int)(t & 127);
  long bn = t >> 7;
  int  b  = (int)(bn >> 13);
  long n  = bn & 8191;
  int iter = cc >> 5, f = cc & 31;
  float v = outs[(long)iter * NN * CF + n * CF + b * 32 + f];
  out[t] = fmaxf(v * scale[cc] + bias[cc], 0.0f);
}

// ------------------------------------------------------------------ launch --
static inline size_t alignUp(size_t v, size_t a) { return (v + a - 1) & ~(a - 1); }

extern "C" void kernel_launch(void* const* d_in, const int* in_sizes, int n_in,
                              void* d_out, int out_size, void* d_ws, size_t ws_size,
                              hipStream_t stream) {
  const float* x     = (const float*)d_in[0];
  const float* x0    = (const float*)d_in[1];
  const float* adj   = (const float*)d_in[2];
  const float* W     = (const float*)d_in[3];
  const float* W0    = (const float*)d_in[4];
  const float* gamma = (const float*)d_in[5];
  const float* beta  = (const float*)d_in[6];
  float*       out   = (float*)d_out;

  // workspace carve-up (~162 MB total)
  char* ws = (char*)d_ws;
  size_t off = 0;
  _Float16* adjh = (_Float16*)(ws + off); off = alignUp(off + (size_t)NN * NN * 2, 256);
  _Float16* hkT  = (_Float16*)(ws + off); off = alignUp(off + (size_t)CF * NN * 2, 256);
  float*    h0f  = (float*)   (ws + off); off = alignUp(off + (size_t)NN * CF * 4, 256);
  float*    outs = (float*)   (ws + off); off = alignUp(off + (size_t)4 * NN * CF * 4, 256);
  float*    part = (float*)   (ws + off); off = alignUp(off + (size_t)KP * NN * CF * 4, 256);
  float*    psum = (float*)   (ws + off); off = alignUp(off + 128 * 64 * 4, 256);
  float*    psq  = (float*)   (ws + off); off = alignUp(off + 128 * 64 * 4, 256);
  float*    scl  = (float*)   (ws + off); off = alignUp(off + 128 * 4, 256);
  float*    bia  = (float*)   (ws + off); off = alignUp(off + 128 * 4, 256);
  (void)ws_size; (void)in_sizes; (void)n_in; (void)out_size;

  // 1) adj f32 -> f16
  k_convert<<<(NN * (long)NN) / 4 / 256, 256, 0, stream>>>(adj, adjh);
  // 2) projections, batch-packed
  k_proj<<<(NN * CF) / 256, 256, 0, stream>>>(x, x0, W, W0, hkT, h0f);
  // 3) four propagation iterations (adj streamed once per iteration)
  for (int iter = 0; iter < 4; ++iter) {
    k_prop<<<256, 256, 0, stream>>>(adjh, hkT, part);          // 2048 waves
    k_reduce_update<<<(NN * CF) / 256, 256, 0, stream>>>(
        part, h0f, outs + (long)iter * NN * CF, hkT);
  }
  // 4) BatchNorm statistics (deterministic two-stage)
  k_bn_partial<<<dim3(128, 64), 128, 0, stream>>>(outs, psum, psq);
  k_bn_final<<<1, 128, 0, stream>>>(psum, psq, gamma, beta, scl, bia);
  // 5) normalize + ReLU into [B,N,128]
  k_finalize<<<(2L * NN * 128) / 256, 256, 0, stream>>>(outs, scl, bia, out);
}
